// MirrorLM_14199161880586
// MI455X (gfx1250) — compile-verified, matrix operand order checked by reading, not run on hardware
//
#include <hip/hip_runtime.h>
#include <stdint.h>

// ---------------- problem constants ----------------
#define B_   32
#define T_   2048
#define V_   50257
#define VPAD 50272            // V padded to multiple of 16 for WMMA N-tiles
#define D_   512
#define H_   512
#define NH_  2
#define HD_  256
#define NROWS (B_ * T_)       // 65536

typedef __attribute__((ext_vector_type(16))) _Float16 v16h;
typedef __attribute__((ext_vector_type(8)))  _Float16 v8h;
typedef __attribute__((ext_vector_type(8)))  float    v8f;
typedef __attribute__((ext_vector_type(4)))  unsigned int u32x4;
typedef __attribute__((ext_vector_type(8)))  int      i32x8;
typedef __attribute__((ext_vector_type(4)))  int      i32x4;

#if defined(__gfx1250__) && __has_builtin(__builtin_amdgcn_tensor_load_to_lds) && \
    __has_builtin(__builtin_amdgcn_s_wait_tensorcnt)
#define HAVE_TDM 1
#else
#define HAVE_TDM 0
#endif

// Load a 16x32-f16 WMMA fragment row-chunk:
// caller bakes row (lane&15) into the pointer; lane>>4 selects the 8-half K
// group; halves [k..k+7] and [k+16..k+23] per the CDNA5 16-bit A/B layout.
__device__ __forceinline__ v16h ldfrag(const _Float16* p) {
  v8h lo = *(const v8h*)(p);
  v8h hi = *(const v8h*)(p + 16);
  return __builtin_shufflevector(lo, hi, 0,1,2,3,4,5,6,7,8,9,10,11,12,13,14,15);
}

__device__ __forceinline__ float phi_f(float x) {   // elu(x)+1
  return x > 0.f ? x + 1.f : expf(x);
}
__device__ __forceinline__ float sigm_f(float x) {
  return 1.f / (1.f + expf(-x));
}

#if HAVE_TDM
// Tensor Data Mover: 2D tile (tileH rows x 64B) global -> LDS, with LDS padding
// of 16B per 64B row so LDS row stride is 80B (bank-conflict-free fragments).
// D# packing per CDNA5 ISA.
// This toolchain's builtin takes 6 args: (g0, g1, g2, g3, extra, cpol).
__device__ __forceinline__ void tdm_load_2d(unsigned lds_off, const void* gptr,
                                            unsigned strideRow4,  // 4B elems/row
                                            unsigned tileH) {
  unsigned long long ga = (unsigned long long)(uintptr_t)gptr;
  u32x4 g0;
  g0[0] = 1u;                                            // count=1 (valid, user)
  g0[1] = lds_off;                                       // lds_addr
  g0[2] = (unsigned)ga;                                  // global_addr[95:64]
  g0[3] = ((unsigned)(ga >> 32) & 0x01FFFFFFu)           // global_addr[120:96]
          | 0x80000000u;                                 // type=2 ("image")
  i32x8 g1;
  unsigned dim0 = strideRow4;                            // tensor_dim0 (elems)
  unsigned dim1 = tileH;                                 // tensor_dim1 (rows)
  // data_size=4B (2), pad_enable, pad_interval=16DW (code 3), pad_amount=4DW (code 3)
  g1[0] = (int)((2u << 16) | (1u << 20) | (3u << 22) | (3u << 25));
  g1[1] = (int)((dim0 & 0xFFFFu) << 16);                 // tensor_dim0[15:0]
  g1[2] = (int)(((dim0 >> 16) & 0xFFFFu) | ((dim1 & 0xFFFFu) << 16));
  g1[3] = (int)(((dim1 >> 16) & 0xFFFFu) | (16u << 16)); // tile_dim0 = 16 elems (64B)
  g1[4] = (int)(tileH & 0xFFFFu);                        // tile_dim1 (tile_dim2=0)
  g1[5] = (int)strideRow4;                               // tensor_dim0_stride[31:0]
  g1[6] = 0;
  g1[7] = 0;
  i32x4 z4 = {0, 0, 0, 0};
  i32x8 z8 = {0, 0, 0, 0, 0, 0, 0, 0};
  __builtin_amdgcn_tensor_load_to_lds(g0, g1, z4, z4, z8, 0);
}
#endif

// ---------------- weight packing ----------------
__global__ void pack_w_f16(const float* __restrict__ src, _Float16* __restrict__ dst,
                           int srcRows, int dstRows, int K) {
  long long idx = (long long)blockIdx.x * blockDim.x + threadIdx.x;
  long long total = (long long)dstRows * K;
  if (idx >= total) return;
  int r = (int)(idx / K);
  float v = (r < srcRows) ? src[idx] : 0.f;
  dst[idx] = (_Float16)v;
}

// transposed pack (for h @ fast_W: B[k][n] = fast_W[k][n] -> Bp[n][k])
__global__ void pack_w_f16_t(const float* __restrict__ src, _Float16* __restrict__ dst,
                             int N, int K) {
  long long idx = (long long)blockIdx.x * blockDim.x + threadIdx.x;
  long long total = (long long)N * K;
  if (idx >= total) return;
  int r = (int)(idx / K);
  int c = (int)(idx % K);
  dst[idx] = (_Float16)src[(long long)c * N + r];
}

// ---------------- embedding gather ----------------
__global__ void embed_kernel(const int* __restrict__ ids, const float* __restrict__ emb,
                             float* __restrict__ x32, _Float16* __restrict__ x16) {
  long long idx = (long long)blockIdx.x * blockDim.x + threadIdx.x;
  long long total = (long long)NROWS * D_;
  if (idx >= total) return;
  int row = (int)(idx >> 9);
  int col = (int)(idx & 511);
  int tok = ids[row];
  float v = emb[(long long)tok * D_ + col];
  x32[idx] = v;
  x16[idx] = (_Float16)v;
}

// ---------------- tiled WMMA GEMM with async (TDM) LDS staging ----------------
// C[M,N] = A[M,K] (f16 row-major) * B, Bp[n][k] = B[k][n] (f16 row-major [N,K]).
// Block = 128x64 C tile, 8 waves, each wave a 32x32 region (2x2 WMMA tiles).
// LDS rows padded to 80B (ASTR=40 halves): lanes 0..15 of a b128 fragment read
// land on 16 disjoint 4-bank groups -> conflict-free.
#define BM   128
#define BN   64
#define BK   32
#define ASTR 40

__global__ void gemm_tiled_wmma(const _Float16* __restrict__ A,
                                const _Float16* __restrict__ Bp,
                                float* __restrict__ C,
                                int Mblocks, int K, int ldc) {
  __shared__ _Float16 Ab[2][BM * ASTR];
  __shared__ _Float16 Bb[2][BN * ASTR];
  int bid = blockIdx.x;
  int mb = bid % Mblocks;
  int nb = bid / Mblocks;
  int rowBase = mb * BM;
  int colBase = nb * BN;
  int tid  = threadIdx.x;
  int lane = tid & 31;
  int wave = tid >> 5;
  int mw = wave & 3;                 // 4 M sub-blocks of 32 rows
  int nw = wave >> 2;                // 2 N sub-blocks of 32 cols
  int half8 = (lane >> 4) << 3;
  int nK = K / BK;

  auto issue_tile = [&](int buf, int k0) {
#if HAVE_TDM
    if (wave == 0) {
      tdm_load_2d((unsigned)(uintptr_t)(void*)&Ab[buf][0],
                  A + (size_t)rowBase * K + k0, (unsigned)(K >> 1), BM);
      tdm_load_2d((unsigned)(uintptr_t)(void*)&Bb[buf][0],
                  Bp + (size_t)colBase * K + k0, (unsigned)(K >> 1), BN);
    }
#else
    {   // A: 128 rows x 2 chunks of 16 halves, one chunk per thread
      int r = tid >> 1, h = tid & 1;
      const _Float16* s = A + (size_t)(rowBase + r) * K + k0 + h * 16;
      _Float16* d = &Ab[buf][r * ASTR + h * 16];
      *(v8h*)(d)     = *(const v8h*)(s);
      *(v8h*)(d + 8) = *(const v8h*)(s + 8);
    }
    if (tid < 128) {   // B: 64 rows x 2 chunks
      int r = tid >> 1, h = tid & 1;
      const _Float16* s = Bp + (size_t)(colBase + r) * K + k0 + h * 16;
      _Float16* d = &Bb[buf][r * ASTR + h * 16];
      *(v8h*)(d)     = *(const v8h*)(s);
      *(v8h*)(d + 8) = *(const v8h*)(s + 8);
    }
#endif
  };

  issue_tile(0, 0);
#if HAVE_TDM
  if (wave == 0) __builtin_amdgcn_s_wait_tensorcnt(0);
#endif
  __syncthreads();

  v8f acc00 = {0.f,0.f,0.f,0.f,0.f,0.f,0.f,0.f};
  v8f acc01 = acc00, acc10 = acc00, acc11 = acc00;

  for (int kk = 0; kk < nK; ++kk) {
    int cur = kk & 1;
    if (kk + 1 < nK) issue_tile(cur ^ 1, (kk + 1) * BK);
    const _Float16* As = &Ab[cur][(mw * 32 + (lane & 15)) * ASTR + half8];
    const _Float16* Bs = &Bb[cur][(nw * 32 + (lane & 15)) * ASTR + half8];
    v16h a0 = ldfrag(As);
    v16h a1 = ldfrag(As + 16 * ASTR);
    v16h b0 = ldfrag(Bs);
    v16h b1 = ldfrag(Bs + 16 * ASTR);
    acc00 = __builtin_amdgcn_wmma_f32_16x16x32_f16(false, a0, false, b0, (short)0, acc00, false, false);
    acc01 = __builtin_amdgcn_wmma_f32_16x16x32_f16(false, a0, false, b1, (short)0, acc01, false, false);
    acc10 = __builtin_amdgcn_wmma_f32_16x16x32_f16(false, a1, false, b0, (short)0, acc10, false, false);
    acc11 = __builtin_amdgcn_wmma_f32_16x16x32_f16(false, a1, false, b1, (short)0, acc11, false, false);
#if HAVE_TDM
    if (wave == 0) __builtin_amdgcn_s_wait_tensorcnt(0);
#endif
    __syncthreads();
  }

  int crow = rowBase + mw * 32;
  int ccol = colBase + nw * 32;
#pragma unroll
  for (int i = 0; i < 8; ++i) {
    int m0 = crow + i + half8;
    int n0 = ccol + (lane & 15);
    C[(size_t)m0 * ldc + n0]              = acc00[i];
    C[(size_t)m0 * ldc + n0 + 16]         = acc01[i];
    C[(size_t)(m0 + 16) * ldc + n0]       = acc10[i];
    C[(size_t)(m0 + 16) * ldc + n0 + 16]  = acc11[i];
  }
}

// ---------------- attention: k_sum / kv_sum over T ----------------
// Y1 layout per row (2048 cols): [q | k | v | g_pre], each 512 = NH*HD.
__global__ void attn_stats_kernel(const float* __restrict__ Y1,
                                  float* __restrict__ ksum, float* __restrict__ kvsum) {
  int bh = blockIdx.x;              // b*NH + nh, 64 blocks
  int b  = bh >> 1, nh = bh & 1;
  int hd = threadIdx.x;             // 256 threads
  const float* base = Y1 + (long long)b * T_ * 2048;
  float sk = 0.f, skv = 0.f;
  for (int t = 0; t < T_; ++t) {
    const float* rowp = base + (long long)t * 2048;
    float kk = phi_f(rowp[512  + nh * HD_ + hd]);
    float vv =       rowp[1024 + nh * HD_ + hd];
    sk  += kk;
    skv += kk * vv;
  }
  ksum [bh * HD_ + hd] = sk;
  kvsum[bh * HD_ + hd] = skv;
}

// ---------------- attention output + gate mix (overwrites x/x16) ----------------
__global__ void attn_out_kernel(const float* __restrict__ Y1,
                                const float* __restrict__ ksum,
                                const float* __restrict__ kvsum,
                                const float* __restrict__ bg,
                                float* __restrict__ x32, _Float16* __restrict__ x16) {
  __shared__ float red[4][256];
  long long row = blockIdx.x;            // 65536 rows
  int tid = threadIdx.x;                 // 256 threads = one hd slot per head
  int b = (int)(row >> 11);              // row / T_
  const float* rowp = Y1 + row * 2048;
  const float* ks0  = ksum  + (b * 2 + 0) * HD_;
  const float* kv0  = kvsum + (b * 2 + 0) * HD_;
  const float* ks1  = ksum  + (b * 2 + 1) * HD_;
  const float* kv1  = kvsum + (b * 2 + 1) * HD_;
  float q0 = phi_f(rowp[tid]);
  float q1 = phi_f(rowp[HD_ + tid]);
  red[0][tid] = q0 * kv0[tid];
  red[1][tid] = q0 * ks0[tid];
  red[2][tid] = q1 * kv1[tid];
  red[3][tid] = q1 * ks1[tid];
  __syncthreads();
  for (int s = 128; s > 0; s >>= 1) {
    if (tid < s) {
#pragma unroll
      for (int i = 0; i < 4; ++i) red[i][tid] += red[i][tid + s];
    }
    __syncthreads();
  }
  float out0 = red[0][0] / (red[1][0] + 1e-6f);
  float out1 = red[2][0] / (red[3][0] + 1e-6f);
#pragma unroll
  for (int half = 0; half < 2; ++half) {
    int d = half * HD_ + tid;
    float o  = half ? out1 : out0;
    float g  = sigm_f(rowp[1536 + d] + bg[d]);
    long long xi = row * D_ + d;
    float xv = x32[xi];
    float nv = g * o + (1.f - g) * xv;
    x32[xi] = nv;
    x16[xi] = (_Float16)nv;
  }
}

// ---------------- persistent recurrent scan ----------------
// Single 1024-thread workgroup (32 waves). h kept as per-thread fp32 registers
// (thread owns column col=tid&511 for 16 batch rows) + f16 mirror in LDS as the
// WMMA A matrix. Per step: P[32,1536] = h @ [W_h.T | W_zh.T | fast_W] via WMMA,
// then the elementwise GRU update. Also emits pred_error_norm.
__global__ void scan_kernel(const float* __restrict__ Y2,        // [NROWS,1024] u_x|z_x
                            const _Float16* __restrict__ Wscan,  // [1536,512] packed
                            const float* __restrict__ b_in,
                            const float* __restrict__ b_zx,
                            const float* __restrict__ log_tau,
                            const float* __restrict__ log_dt,
                            float* __restrict__ h_last,          // [32,512]
                            float* __restrict__ norm_out) {      // scalar
  extern __shared__ char smem[];
  _Float16* h16 = (_Float16*)smem;                              // 32*512 f16 = 32 KB
  float*    P   = (float*)(smem + 32 * 512 * sizeof(_Float16)); // 32*1536 f32 = 192 KB
  int tid  = threadIdx.x;
  int lane = tid & 31, wave = tid >> 5;
  int col  = tid & 511;
  float bin_c = b_in[col], bzx_c = b_zx[col];
  float tau_c = expf(log_tau[col]), dt_c = expf(log_dt[col]);
  float hreg[16], hprev[16];
#pragma unroll
  for (int r = 0; r < 16; ++r) { hreg[r] = 0.f; hprev[r] = 0.f; }
#pragma unroll
  for (int r = 0; r < 16; ++r) {
    int b = (tid >> 9) + (r << 1);
    h16[b * 512 + col] = (_Float16)0.f;
  }
  __syncthreads();

  int mt     = wave & 1;
  int ntBase = (wave >> 1) * 6;   // 32 waves * 6 tiles = 192 = 2 x 96 tiles
  int half8  = (lane >> 4) << 3;

  for (int t = 0; t < T_; ++t) {
    for (int j = 0; j < 6; ++j) {
      int nt = ntBase + j;
      const _Float16* pa = h16   + (mt * 16 + (lane & 15)) * 512 + half8;
      const _Float16* pb = Wscan + (long long)(nt * 16 + (lane & 15)) * 512 + half8;
      v8f acc = {0.f,0.f,0.f,0.f,0.f,0.f,0.f,0.f};
#pragma unroll
      for (int kk = 0; kk < 16; ++kk) {
        v16h a = ldfrag(pa + kk * 32);
        v16h b = ldfrag(pb + kk * 32);
        acc = __builtin_amdgcn_wmma_f32_16x16x32_f16(false, a, false, b, (short)0, acc,
                                                     false, false);
      }
      int n = nt * 16 + (lane & 15);
#pragma unroll
      for (int i = 0; i < 8; ++i) {
        int m = mt * 16 + i + half8;
        P[m * 1536 + n] = acc[i];
      }
    }
    __syncthreads();
#pragma unroll
    for (int r = 0; r < 16; ++r) {
      int b = (tid >> 9) + (r << 1);
      long long rowo = ((long long)b * T_ + t) * 1024;
      float h  = hreg[r];
      float u  = Y2[rowo + col] + bin_c + P[b * 1536 + col];
      float f  = tanhf(u);
      float hl = h + dt_c * (f - tau_c * h);
      float z  = sigm_f(Y2[rowo + 512 + col] + bzx_c + P[b * 1536 + 512 + col]);
      float sl = (1.f - z) * h + z * hl;
      float hn = sl + P[b * 1536 + 1024 + col];
      hprev[r] = h;
      hreg[r]  = hn;
      h16[b * 512 + col] = (_Float16)hn;
    }
    __syncthreads();
  }

  float sq = 0.f;
#pragma unroll
  for (int r = 0; r < 16; ++r) {
    int b = (tid >> 9) + (r << 1);
    h_last[b * 512 + col] = hreg[r];
    float d = hreg[r] - hprev[r];
    sq += d * d;
  }
  P[tid] = sq;
  __syncthreads();
  for (int s = 512; s > 0; s >>= 1) {
    if (tid < s) P[tid] += P[tid + s];
    __syncthreads();
  }
  if (tid == 0) norm_out[0] = sqrtf(P[0] / (float)(B_ * H_));
}

// ---------------- LayerNorm ----------------
__global__ void layernorm_kernel(const float* __restrict__ hlast,
                                 const float* __restrict__ g, const float* __restrict__ bb,
                                 float* __restrict__ outh, _Float16* __restrict__ h16) {
  __shared__ float red[512];
  __shared__ float s_mu, s_rstd;
  int r = blockIdx.x, c = threadIdx.x;    // 32 blocks x 512 threads
  float v = hlast[r * 512 + c];
  red[c] = v;
  __syncthreads();
  for (int s = 256; s > 0; s >>= 1) { if (c < s) red[c] += red[c + s]; __syncthreads(); }
  if (c == 0) s_mu = red[0] / 512.f;
  __syncthreads();
  float dv = v - s_mu;
  red[c] = dv * dv;
  __syncthreads();
  for (int s = 256; s > 0; s >>= 1) { if (c < s) red[c] += red[c + s]; __syncthreads(); }
  if (c == 0) s_rstd = rsqrtf(red[0] / 512.f + 1e-5f);
  __syncthreads();
  float o = dv * s_rstd * g[c] + bb[c];
  outh[r * 512 + c] = o;
  h16[r * 512 + c]  = (_Float16)o;
}

// ---------------- LM head GEMM (WMMA, bias + N-bounds epilogue) ----------------
__global__ void gemm_lm_wmma(const _Float16* __restrict__ A,     // [32,512]
                             const _Float16* __restrict__ Bp,    // [VPAD,512]
                             const float* __restrict__ blm,
                             float* __restrict__ out) {          // [32,V_]
  int wave = blockIdx.x * (blockDim.x >> 5) + (threadIdx.x >> 5);
  const int Ntiles = VPAD / 16;
  if (wave >= 2 * Ntiles) return;
  int lane  = threadIdx.x & 31;
  int mt    = wave & 1;
  int nt    = wave >> 1;
  int half8 = (lane >> 4) << 3;
  const _Float16* pa = A  + (long long)(mt * 16 + (lane & 15)) * 512 + half8;
  const _Float16* pb = Bp + (long long)(nt * 16 + (lane & 15)) * 512 + half8;
  v8f acc = {0.f,0.f,0.f,0.f,0.f,0.f,0.f,0.f};
#pragma unroll
  for (int kk = 0; kk < 16; ++kk) {
    __builtin_prefetch(pb + kk * 32 + 128, 0, 1);
    v16h a = ldfrag(pa + kk * 32);
    v16h b = ldfrag(pb + kk * 32);
    acc = __builtin_amdgcn_wmma_f32_16x16x32_f16(false, a, false, b, (short)0, acc,
                                                 false, false);
  }
  int n = nt * 16 + (lane & 15);
  if (n < V_) {
    float bias = blm[n];
#pragma unroll
    for (int i = 0; i < 8; ++i) {
      int m = mt * 16 + i + half8;
      out[(long long)m * V_ + n] = acc[i] + bias;
    }
  }
}

// ---------------- host launch ----------------
extern "C" void kernel_launch(void* const* d_in, const int* in_sizes, int n_in,
                              void* d_out, int out_size, void* d_ws, size_t ws_size,
                              hipStream_t stream) {
  (void)in_sizes; (void)n_in; (void)out_size; (void)ws_size;
  const int*   ids  = (const int*)  d_in[0];
  const float* emb  = (const float*)d_in[1];
  const float* Wq   = (const float*)d_in[2];
  const float* Wk   = (const float*)d_in[3];
  const float* Wv   = (const float*)d_in[4];
  /* d_in[5] = Wo, unused in forward */
  const float* Wg   = (const float*)d_in[6];
  const float* bg   = (const float*)d_in[7];
  const float* W_in = (const float*)d_in[8];
  const float* b_in = (const float*)d_in[9];
  const float* W_h  = (const float*)d_in[10];
  const float* ltau = (const float*)d_in[11];
  const float* ldt  = (const float*)d_in[12];
  const float* W_zx = (const float*)d_in[13];
  const float* b_zx = (const float*)d_in[14];
  const float* W_zh = (const float*)d_in[15];
  const float* fW   = (const float*)d_in[16];
  const float* ln_g = (const float*)d_in[17];
  const float* ln_b = (const float*)d_in[18];
  const float* Wlm  = (const float*)d_in[19];
  const float* blm  = (const float*)d_in[20];
  float* out = (float*)d_out;
  float* out_h    = out + (long long)B_ * V_;       // h output section
  float* out_norm = out_h + (long long)B_ * H_;     // scalar section

  char* ws = (char*)d_ws;
  size_t off = 0;
  auto take = [&](size_t bytes) -> char* {
    char* p = ws + off;
    off = (off + bytes + 255) & ~(size_t)255;
    return p;
  };
  float*    X32   = (float*)   take((size_t)NROWS * D_ * 4);
  _Float16* X16   = (_Float16*)take((size_t)NROWS * D_ * 2);
  float*    Y1    = (float*)   take((size_t)NROWS * 2048 * 4);
  float*    Y2    = (float*)   take((size_t)NROWS * 1024 * 4);
  _Float16* WP1   = (_Float16*)take((size_t)2048 * 512 * 2);
  _Float16* WP2   = (_Float16*)take((size_t)1024 * 512 * 2);
  _Float16* WPS   = (_Float16*)take((size_t)1536 * 512 * 2);
  _Float16* WPL   = (_Float16*)take((size_t)VPAD * 512 * 2);
  float*    KS    = (float*)   take((size_t)64 * HD_ * 4);
  float*    KVS   = (float*)   take((size_t)64 * HD_ * 4);
  float*    HLAST = (float*)   take((size_t)B_ * H_ * 4);
  _Float16* H16LM = (_Float16*)take((size_t)B_ * H_ * 2);

  auto g1 = [](long long n, int bs) { return dim3((unsigned)((n + bs - 1) / bs)); };

  // 1) pack weights -> f16 B-operand layout
  long long wsz = 512LL * 512;
  pack_w_f16<<<g1(wsz,256),256,0,stream>>>(Wq,   WP1 + 0*512*512, 512, 512, 512);
  pack_w_f16<<<g1(wsz,256),256,0,stream>>>(Wk,   WP1 + 1*512*512, 512, 512, 512);
  pack_w_f16<<<g1(wsz,256),256,0,stream>>>(Wv,   WP1 + 2*512*512, 512, 512, 512);
  pack_w_f16<<<g1(wsz,256),256,0,stream>>>(Wg,   WP1 + 3*512*512, 512, 512, 512);
  pack_w_f16<<<g1(wsz,256),256,0,stream>>>(W_in, WP2 + 0*512*512, 512, 512, 512);
  pack_w_f16<<<g1(wsz,256),256,0,stream>>>(W_zx, WP2 + 1*512*512, 512, 512, 512);
  pack_w_f16<<<g1(wsz,256),256,0,stream>>>(W_h,  WPS + 0*512*512, 512, 512, 512);
  pack_w_f16<<<g1(wsz,256),256,0,stream>>>(W_zh, WPS + 1*512*512, 512, 512, 512);
  pack_w_f16_t<<<g1(wsz,256),256,0,stream>>>(fW, WPS + 2*512*512, 512, 512);
  pack_w_f16<<<g1((long long)VPAD*512,256),256,0,stream>>>(Wlm, WPL, V_, VPAD, 512);

  // 2) embedding gather
  embed_kernel<<<g1((long long)NROWS * D_, 256), 256, 0, stream>>>(ids, emb, X32, X16);

  // 3) fused q/k/v/gate projections: Y1[65536,2048] = X16 @ WP1^T (tiled+TDM)
  {
    int Mblocks = NROWS / BM;                   // 512
    int Nblocks = 2048 / BN;                    // 32
    gemm_tiled_wmma<<<dim3(Mblocks * Nblocks), 256, 0, stream>>>(X16, WP1, Y1,
                                                                 Mblocks, 512, 2048);
  }

  // 4) attention reductions over T
  attn_stats_kernel<<<dim3(B_ * NH_), HD_, 0, stream>>>(Y1, KS, KVS);

  // 5) attention output + gate mix (overwrites X32/X16 with post-attn x)
  attn_out_kernel<<<dim3(NROWS), HD_, 0, stream>>>(Y1, KS, KVS, bg, X32, X16);

  // 6) hoisted GRU input projections: Y2[65536,1024] = X16 @ [W_in|W_zx]^T
  {
    int Mblocks = NROWS / BM;                   // 512
    int Nblocks = 1024 / BN;                    // 16
    gemm_tiled_wmma<<<dim3(Mblocks * Nblocks), 256, 0, stream>>>(X16, WP2, Y2,
                                                                 Mblocks, 512, 1024);
  }

  // 7) persistent serial scan (single WGP-resident workgroup, WMMA inner GEMM)
  {
    size_t smem = (size_t)32 * 512 * sizeof(_Float16) + (size_t)32 * 1536 * sizeof(float);
    scan_kernel<<<dim3(1), 1024, smem, stream>>>(Y2, WPS, b_in, b_zx, ltau, ldt,
                                                 HLAST, out_norm);
  }

  // 8) LayerNorm -> h output + f16 copy for LM head
  layernorm_kernel<<<dim3(B_), H_, 0, stream>>>(HLAST, ln_g, ln_b, out_h, H16LM);

  // 9) LM head: logits[32,50257] = h16 @ WPL^T + blm
  {
    long long tiles = 2LL * (VPAD / 16);
    gemm_lm_wmma<<<g1(tiles, 8), 256, 0, stream>>>(H16LM, WPL, blm, out);
  }
}
